// FALoss_v2_57501022159376
// MI455X (gfx1250) — matrix operand order repaired and checked by reading, not compiled, not run on hardware
//
#include <hip/hip_runtime.h>

typedef __attribute__((ext_vector_type(16))) __bf16 v16bf;
typedef __attribute__((ext_vector_type(8)))  __bf16 v8bf;
typedef __attribute__((ext_vector_type(8)))  float  v8f;

#define CCH 512
#define KHW 16384
#define BT  128     // output tile per block (each dim)
#define KB  64      // K slab per LDS stage
#define LDP 72      // padded LDS row stride in bf16 elements (64 + 8 -> conflict-free)

__device__ __forceinline__ unsigned pack2bf(float lo, float hi) {
    // round-to-nearest-even f32 -> bf16, packed pair
    unsigned a = __float_as_uint(lo), b = __float_as_uint(hi);
    a += 0x7FFFu + ((a >> 16) & 1u);
    b += 0x7FFFu + ((b >> 16) & 1u);
    return (a >> 16) | (b & 0xFFFF0000u);
}

// One block -> one 128x128 tile (ti<=tj) of one gram; mirrors the transpose.
// 8 waves, each wave 2x4 wmma tiles (32x64). Next K-slab is prefetched into
// VGPRs before the wmma phase so HBM latency overlaps matrix math.
__global__ __launch_bounds__(256, 1)
void gram_wmma_kernel(const float* __restrict__ det,
                      const float* __restrict__ sr,
                      float* __restrict__ grams)
{
    __shared__ __bf16 lA[BT * LDP];
    __shared__ __bf16 lB[BT * LDP];

    const int pair = blockIdx.y;            // 0..15 : 0-7 det batches, 8-15 sr batches
    const int t    = blockIdx.x;            // 0..9  : upper-triangular 4x4 tile index
    const int ti   = (0x0E9500 >> (2 * t)) & 3;   // {0,0,0,0,1,1,1,2,2,3}
    const int tj   = (0xFB9E4  >> (2 * t)) & 3;   // {0,1,2,3,1,2,3,2,3,3}

    const float* __restrict__ src =
        (pair < 8 ? det : sr) + (size_t)(pair & 7) * ((size_t)CCH * KHW);
    float* __restrict__ g = grams + (size_t)pair * (CCH * CCH);

    const int tid  = threadIdx.x;
    const int lane = tid & 31;
    const int wave = tid >> 5;
    const int hh   = lane >> 4;             // half-wave select
    const int l15  = lane & 15;

    const int wm0 = (wave >> 1) * 2;        // m-tile base within block: 0,2,4,6
    const int wn0 = (wave & 1) * 4;         // n-tile base within block: 0,4

    // per-thread staging coordinates (8 float4 per slab)
    const int srow = tid >> 4;              // base row 0..15 (steps of 16 below)
    const int sc4  = (tid & 15) << 2;       // f32 col 0,4,..,60

    v8f acc[2][4];
#pragma unroll
    for (int a = 0; a < 2; ++a)
#pragma unroll
        for (int b = 0; b < 4; ++b) acc[a][b] = (v8f){0.f,0.f,0.f,0.f,0.f,0.f,0.f,0.f};

    float4 pva[8], pvb[8];

    // prologue: prefetch first K slab
#pragma unroll
    for (int i = 0; i < 8; ++i) {
        const int row = srow + i * 16;
        pva[i] = *(const float4*)(src + (size_t)(ti * BT + row) * KHW + sc4);
        pvb[i] = *(const float4*)(src + (size_t)(tj * BT + row) * KHW + sc4);
    }

    for (int k0 = 0; k0 < KHW; k0 += KB) {
        __syncthreads();   // previous compute done before overwriting LDS
        // convert prefetched f32 -> bf16, stage into LDS
#pragma unroll
        for (int i = 0; i < 8; ++i) {
            const int row = srow + i * 16;
            uint2 pa, pb;
            pa.x = pack2bf(pva[i].x, pva[i].y); pa.y = pack2bf(pva[i].z, pva[i].w);
            pb.x = pack2bf(pvb[i].x, pvb[i].y); pb.y = pack2bf(pvb[i].z, pvb[i].w);
            *(uint2*)&lA[row * LDP + sc4] = pa;
            *(uint2*)&lB[row * LDP + sc4] = pb;
        }
        __syncthreads();

        // issue next slab's global loads now; their wait lands after the wmma
        // phase (next iteration's convert), hiding HBM latency.
        const int kn = (k0 + KB < KHW) ? (k0 + KB) : 0;   // last iter: dummy reload
#pragma unroll
        for (int i = 0; i < 8; ++i) {
            const int row = srow + i * 16;
            pva[i] = *(const float4*)(src + (size_t)(ti * BT + row) * KHW + kn + sc4);
            pvb[i] = *(const float4*)(src + (size_t)(tj * BT + row) * KHW + kn + sc4);
        }

#pragma unroll
        for (int ks = 0; ks < KB; ks += 32) {
            v16bf afr[2], bfr[4];
            // A fragment: 16x32 bf16. lanes 0-15: K {0..7, 16..23};
            // lanes 16-31: K {8..15, 24..31}.  (ISA 7.12.2)
#pragma unroll
            for (int a = 0; a < 2; ++a) {
                const int row = (wm0 + a) * 16 + l15;
                const v8bf lo = *(const v8bf*)&lA[row * LDP + ks + hh * 8];
                const v8bf hi = *(const v8bf*)&lA[row * LDP + ks + 16 + hh * 8];
#pragma unroll
                for (int e = 0; e < 8; ++e) { afr[a][e] = lo[e]; afr[a][8 + e] = hi[e]; }
            }
            // B fragment: 32x16 bf16, B[k][n] = F[tjBase+n][k].
            // lanes 0-15: col n, K {0..7, 16..23}; lanes 16-31: K {8..15, 24..31}.
#pragma unroll
            for (int b = 0; b < 4; ++b) {
                const int row = (wn0 + b) * 16 + l15;
                const v8bf lo = *(const v8bf*)&lB[row * LDP + ks + hh * 8];
                const v8bf hi = *(const v8bf*)&lB[row * LDP + ks + 16 + hh * 8];
#pragma unroll
                for (int e = 0; e < 8; ++e) { bfr[b][e] = lo[e]; bfr[b][8 + e] = hi[e]; }
            }
#pragma unroll
            for (int a = 0; a < 2; ++a)
#pragma unroll
                for (int b = 0; b < 4; ++b)
                    acc[a][b] = __builtin_amdgcn_wmma_f32_16x16x32_bf16(
                        false, afr[a], false, bfr[b], (short)0, acc[a][b], false, false);
        }
    }

    // D layout (ISA 7.12.2): VGPR v, lanes 0-15 -> M=v, lanes 16-31 -> M=v+8; N=lane&15.
    const bool mirror = (ti != tj);
#pragma unroll
    for (int a = 0; a < 2; ++a)
#pragma unroll
        for (int b = 0; b < 4; ++b) {
            const int gm = ti * BT + (wm0 + a) * 16 + hh * 8;
            const int gn = tj * BT + (wn0 + b) * 16 + l15;
#pragma unroll
            for (int v = 0; v < 8; ++v) {
                g[(size_t)(gm + v) * CCH + gn] = acc[a][b][v];
                if (mirror) g[(size_t)gn * CCH + (gm + v)] = acc[a][b][v];
            }
        }
}

__global__ void zero_out_kernel(float* out) { out[0] = 0.0f; }

// Normalize both grams by rsqrt(diag_i * diag_j), MSE-reduce into out[0].
__global__ __launch_bounds__(256)
void faloss_reduce_kernel(const float* __restrict__ grams, float* __restrict__ out)
{
    __shared__ float wsum[8];
    const int idx = blockIdx.x * 256 + threadIdx.x;  // 0 .. 8*512*512-1
    const int b   = idx >> 18;
    const int rem = idx & 0x3FFFF;
    const int i   = rem >> 9;
    const int j   = rem & 511;

    const float* gd = grams + (size_t)b * (CCH * CCH);
    const float* gs = grams + (size_t)(8 + b) * (CCH * CCH);

    const float sd = gd[(size_t)i * CCH + j] *
                     rsqrtf(gd[(size_t)i * CCH + i] * gd[(size_t)j * CCH + j]);
    const float ss = gs[(size_t)i * CCH + j] *
                     rsqrtf(gs[(size_t)i * CCH + i] * gs[(size_t)j * CCH + j]);
    const float d = sd - ss;
    float v = d * d;

#pragma unroll
    for (int o = 16; o > 0; o >>= 1) v += __shfl_down(v, o, 32);
    if ((threadIdx.x & 31) == 0) wsum[threadIdx.x >> 5] = v;
    __syncthreads();
    if (threadIdx.x == 0) {
        float s = 0.f;
#pragma unroll
        for (int w = 0; w < 8; ++w) s += wsum[w];
        atomicAdd(out, s * (1.0f / (8.0f * 512.0f * 512.0f)));
    }
}

extern "C" void kernel_launch(void* const* d_in, const int* in_sizes, int n_in,
                              void* d_out, int out_size, void* d_ws, size_t ws_size,
                              hipStream_t stream) {
    const float* det = (const float*)d_in[0];
    const float* sr  = (const float*)d_in[1];
    float* out   = (float*)d_out;
    float* grams = (float*)d_ws;   // 16 * 512 * 512 * 4B = 16 MB

    // 16 grams x 10 upper-triangular tiles; x-major over tiles keeps a pair's
    // 32 MB source hot in the 192 MB L2 across its blocks.
    dim3 ggrid(10, 16);
    gram_wmma_kernel<<<ggrid, 256, 0, stream>>>(det, sr, grams);

    zero_out_kernel<<<1, 1, 0, stream>>>(out);

    const int total = 8 * 512 * 512;
    faloss_reduce_kernel<<<total / 256, 256, 0, stream>>>(grams, out);
}
